// InteractionBlock_16904991277616
// MI455X (gfx1250) — matrix operand order repaired and checked by reading, not compiled, not run on hardware
//
#include <hip/hip_runtime.h>
#include <hip/hip_bf16.h>

#define NNODES 20000
#define NEDGES 320000

typedef __attribute__((ext_vector_type(16))) _Float16 v16h;
typedef __attribute__((ext_vector_type(8)))  _Float16 v8h;
typedef __attribute__((ext_vector_type(8)))  float    v8f;
typedef __attribute__((ext_vector_type(4)))  unsigned int v4u;
typedef __attribute__((ext_vector_type(8)))  int v8i;
typedef __attribute__((ext_vector_type(4)))  int v4i;

// normalization constants (folded into weights where possible)
#define C_SILU_F     1.6791767f      // 1/sqrt(E[silu(z)^2]), z~N(0,1)
#define INV_SQRT3_F  0.57735027f
#define INV_SQRT2_F  0.70710678f
#define INV_SQRT_C_F 0.125f          // 1/sqrt(64)
#define INV_SQRT_2C_F 0.08838835f    // 1/sqrt(128)
#define INV_SQRT_3C_F 0.07216878f    // 1/sqrt(192)
#define INV_SQRT_NB_F 0.35355339f    // 1/sqrt(8)

// edge-kernel LDS layout (halfs): [0,30720) swizzled Wm1..Wm4, then 8 wave
// staging tiles of 1024 halfs each.  61440 + 16384 = 77824 bytes dynamic LDS.
#define WLDS_HALFS 30720
#define EDGE_LDS_BYTES (WLDS_HALFS * 2 + 8 * 1024 * 2)

__device__ __forceinline__ v8f vzero8() {
  v8f z = {0.f,0.f,0.f,0.f,0.f,0.f,0.f,0.f};
  return z;
}

__device__ __forceinline__ v8f wmma16(v16h a, v16h b, v8f c) {
  // D = A(16x32 f16) * B(32x16 f16) + C(16x16 f32)
  return __builtin_amdgcn_wmma_f32_16x16x32_f16(false, a, false, b, (short)0, c,
                                                false, false);
}

__device__ __forceinline__ v8f silu8(v8f x) {
  v8f r;
#pragma unroll
  for (int i = 0; i < 8; ++i) {
    float t = x[i];
    r[i] = (t / (1.f + __expf(-t))) * C_SILU_F;
  }
  return r;
}

// ---- A operand loaders (ISA 7.12.2: 16-bit A 16x32) -----------------------
// lane L: row = L&15 ; element e holds K = (e>>3)*16 + (L>>4)*8 + (e&7)
// caller passes p = rowBase + k0 + (L>>4)*8  (elements 0..7 at p[0..7],
// elements 8..15 at p[16..23])

__device__ __forceinline__ v16h load_a_f32(const float* __restrict__ p) {
  v16h a;
#pragma unroll
  for (int e = 0; e < 8; ++e) {
    a[e]     = (_Float16)p[e];
    a[8 + e] = (_Float16)p[16 + e];
  }
  return a;
}

// same, but source has element stride 3 (interleaved xyz vectors)
__device__ __forceinline__ v16h load_a_f32_s3(const float* __restrict__ p) {
  v16h a;
#pragma unroll
  for (int e = 0; e < 8; ++e) {
    a[e]     = (_Float16)p[3 * e];
    a[8 + e] = (_Float16)p[3 * (16 + e)];
  }
  return a;
}

// from row-major f16 LDS tile: two contiguous 16-byte ds loads
__device__ __forceinline__ v16h load_a_lds(const _Float16* p) {
  v8h lo = *(const v8h*)p;
  v8h hi = *(const v8h*)(p + 16);
  v16h a;
#pragma unroll
  for (int e = 0; e < 8; ++e) {
    a[e]     = lo[e];
    a[8 + e] = hi[e];
  }
  return a;
}

// ---- B operand from pre-swizzled f16 buffer (global or LDS) ---------------
// layout: [nt][ks][lane][e] ; one contiguous 32B load per operand
__device__ __forceinline__ v16h load_b_sw(const _Float16* __restrict__ w,
                                          int nt, int nk, int ks, int lane) {
  return *(const v16h*)(w + (size_t)((nt * nk + ks) * 32 + lane) * 16);
}

__device__ __forceinline__ v16h load_b_lds(const _Float16* w, int nt, int nk,
                                           int ks, int lane) {
  return *(const v16h*)(w + (size_t)((nt * nk + ks) * 32 + lane) * 16);
}

// ---- weight swizzler: f32 KxN -> f16 B-operand layout, scale folded -------
// B lane mapping (ISA 7.12.2): lane L holds N=L&15, element e holds
// K = ks*32 + (e>>3)*16 + (L>>4)*8 + (e&7).  k >= Kact zero-padded (layer-1 K=8 pad to 32).
__global__ void swizzle_w_kernel(const float* __restrict__ W,
                                 _Float16* __restrict__ dst,
                                 int Kact, int Kpad, int Ncols, float scale) {
  int nk = Kpad >> 5;
  int total = (Ncols >> 4) * nk * 512;
  for (int idx = blockIdx.x * blockDim.x + threadIdx.x; idx < total;
       idx += gridDim.x * blockDim.x) {
    int e    = idx & 15;
    int lane = (idx >> 4) & 31;
    int rem  = idx >> 9;
    int ks   = rem % nk;
    int nt   = rem / nk;
    int k = ks * 32 + ((e >> 3) << 4) + ((lane >> 4) << 3) + (e & 7);
    int n = nt * 16 + (lane & 15);
    float v = (k < Kact) ? W[(size_t)k * Ncols + n] * scale : 0.f;
    dst[idx] = (_Float16)v;
  }
}

__global__ void zero_kernel(float* __restrict__ p, size_t n) {
  size_t i = (size_t)blockIdx.x * blockDim.x + threadIdx.x;
  size_t stride = (size_t)gridDim.x * blockDim.x;
  for (; i < n; i += stride) p[i] = 0.f;
}

// ---------------------------------------------------------------------------
// Kernel 1: node up-projection.  s_up = s @ Wup0/sqrt(C),
// v_up[n,k,m] = sum_c v[n,c,m] * Wup1[c,k] / sqrt(C)
// ---------------------------------------------------------------------------
__global__ void __launch_bounds__(256) node_up_kernel(
    const float* __restrict__ node_feats, const _Float16* __restrict__ w0,
    const _Float16* __restrict__ w1, float* __restrict__ s_up,
    float* __restrict__ v_up) {
  const int tid = threadIdx.x;
  const int wv = tid >> 5, lane = tid & 31;
  const int hf = lane >> 4, l16 = lane & 15;
  const int t = blockIdx.x * 8 + wv;
  if (t >= NNODES / 16) return;
  const int row0 = t * 16;
  const int myrow = row0 + l16;

  // scalar part: A rows from node_feats[:, 0:64]
  {
    const float* rp = node_feats + (size_t)myrow * 256;
    v16h a0 = load_a_f32(rp + 0 + hf * 8);
    v16h a1 = load_a_f32(rp + 32 + hf * 8);
#pragma unroll
    for (int nt = 0; nt < 4; ++nt) {
      v8f c = vzero8();
      c = wmma16(a0, load_b_sw(w0, nt, 2, 0, lane), c);
      c = wmma16(a1, load_b_sw(w0, nt, 2, 1, lane), c);
#pragma unroll
      for (int r = 0; r < 8; ++r)
        s_up[(size_t)(row0 + r + hf * 8) * 64 + nt * 16 + l16] = c[r];
    }
  }
  // vector part: three stride-3 GEMMs
#pragma unroll
  for (int m = 0; m < 3; ++m) {
    const float* rp = node_feats + (size_t)myrow * 256 + 64 + m;
    v16h a0 = load_a_f32_s3(rp + 3 * (0 + hf * 8));
    v16h a1 = load_a_f32_s3(rp + 3 * (32 + hf * 8));
#pragma unroll
    for (int nt = 0; nt < 4; ++nt) {
      v8f c = vzero8();
      c = wmma16(a0, load_b_sw(w1, nt, 2, 0, lane), c);
      c = wmma16(a1, load_b_sw(w1, nt, 2, 1, lane), c);
#pragma unroll
      for (int r = 0; r < 8; ++r)
        v_up[(size_t)(row0 + r + hf * 8) * 192 + (nt * 16 + l16) * 3 + m] = c[r];
    }
  }
}

// ---------------------------------------------------------------------------
// Kernel 2: fused edge MLP (4 WMMA layers) + message build + atomic scatter.
// One wave owns 16 edges.  All four weight matrices (61440 B, pre-swizzled
// into B-operand layout) are DMA'd into LDS once per block by the Tensor Data
// Mover; every WMMA B operand is then two ds_load_b128s instead of GL2 trips.
// Intermediate activations relayout via per-wave LDS staging tile.
// ---------------------------------------------------------------------------
__global__ void __launch_bounds__(256) edge_kernel(
    const float* __restrict__ edge_feats, const float* __restrict__ edge_attrs,
    const int* __restrict__ iedges, const int* __restrict__ jedges,
    const _Float16* __restrict__ wg,  // base of contiguous Wm1|Wm2|Wm3|Wm4
    const float* __restrict__ s_up, const float* __restrict__ v_up,
    float* __restrict__ msg_s, float* __restrict__ msg_v) {
  extern __shared__ _Float16 smem[];
  const int tid = threadIdx.x;
  const int wv = tid >> 5, lane = tid & 31;
  const int hf = lane >> 4, l16 = lane & 15;
  const int e0 = (blockIdx.x * 8 + wv) * 16;

  const _Float16* w1 = smem;          // 2048 halfs  (8->64, K padded to 32)
  const _Float16* w2 = smem + 2048;   // 4096 halfs
  const _Float16* w3 = smem + 6144;   // 4096 halfs
  const _Float16* w4 = smem + 10240;  // 20480 halfs
  _Float16* hs = smem + WLDS_HALFS + wv * 1024;

  // ---- TDM: stage all edge-MLP weights into LDS (one 61440B 1-D tile) ----
  if (wv == 0) {
    unsigned lds_addr = (unsigned)(uintptr_t)(void*)smem;  // low32 = LDS offset
    unsigned long long ga = (unsigned long long)(uintptr_t)wg;
    v4u g0;
    g0[0] = 1u;                                   // count=1, user mode
    g0[1] = lds_addr;                             // D#.lds_addr
    g0[2] = (unsigned)(ga & 0xffffffffu);         // global_addr[31:0]
    g0[3] = (unsigned)((ga >> 32) & 0x01ffffffu)  // global_addr[56:32]
            | (2u << 30);                         // type=2 ("image")
    v8i g1;
    g1[0] = (int)(3u << 16);        // workgroup_mask=0, data_size=3 (8B)
    g1[1] = (int)(7680u << 16);     // tensor_dim0[15:0] in [31:16]
    g1[2] = (int)(7680u >> 16);     // tensor_dim0[31:16] ; tensor_dim1 lo16=0
    g1[3] = (int)(7680u << 16);     // tile_dim0 = 7680 in [31:16]
    g1[4] = 0;                      // tile_dim1=0 (unused), tile_dim2=0
    g1[5] = (int)7680u;             // tensor_dim0_stride[31:0]
    g1[6] = 0;                      // stride hi / dim1 stride lo
    g1[7] = 0;
    v4i z4 = {0, 0, 0, 0};
#if defined(__clang_major__) && __clang_major__ >= 23
    v8i z8 = {0, 0, 0, 0, 0, 0, 0, 0};
    __builtin_amdgcn_tensor_load_to_lds(g0, g1, z4, z4, z8, 0);
#else
    __builtin_amdgcn_tensor_load_to_lds(g0, g1, z4, z4, 0);
#endif
    __builtin_amdgcn_s_wait_tensorcnt(0);
  }
  __syncthreads();

  v8f acc[4];

  // ---- layer 1: edge_feats (E x 8) @ Wm1 (8->64), K zero-padded to 32 ----
  {
    v16h a;
#pragma unroll
    for (int e = 0; e < 16; ++e) a[e] = (_Float16)0.f;
    if (hf == 0) {  // only K=0..7 lanes carry real data
      const float* src = edge_feats + (size_t)(e0 + l16) * 8;
#pragma unroll
      for (int e = 0; e < 8; ++e) a[e] = (_Float16)src[e];
    }
#pragma unroll
    for (int nt = 0; nt < 4; ++nt) {
      v8f c = vzero8();
      c = wmma16(a, load_b_lds(w1, nt, 1, 0, lane), c);
      acc[nt] = silu8(c);
    }
  }
  // D layout -> row-major f16 staging
#pragma unroll
  for (int nt = 0; nt < 4; ++nt)
#pragma unroll
    for (int r = 0; r < 8; ++r)
      hs[(r + hf * 8) * 64 + nt * 16 + l16] = (_Float16)acc[nt][r];
  __syncthreads();

  // ---- layer 2 ----
  {
    v16h a0 = load_a_lds(hs + l16 * 64 + 0 + hf * 8);
    v16h a1 = load_a_lds(hs + l16 * 64 + 32 + hf * 8);
#pragma unroll
    for (int nt = 0; nt < 4; ++nt) {
      v8f c = vzero8();
      c = wmma16(a0, load_b_lds(w2, nt, 2, 0, lane), c);
      c = wmma16(a1, load_b_lds(w2, nt, 2, 1, lane), c);
      acc[nt] = silu8(c);
    }
  }
  __syncthreads();
#pragma unroll
  for (int nt = 0; nt < 4; ++nt)
#pragma unroll
    for (int r = 0; r < 8; ++r)
      hs[(r + hf * 8) * 64 + nt * 16 + l16] = (_Float16)acc[nt][r];
  __syncthreads();

  // ---- layer 3 ----
  {
    v16h a0 = load_a_lds(hs + l16 * 64 + 0 + hf * 8);
    v16h a1 = load_a_lds(hs + l16 * 64 + 32 + hf * 8);
#pragma unroll
    for (int nt = 0; nt < 4; ++nt) {
      v8f c = vzero8();
      c = wmma16(a0, load_b_lds(w3, nt, 2, 0, lane), c);
      c = wmma16(a1, load_b_lds(w3, nt, 2, 1, lane), c);
      acc[nt] = silu8(c);
    }
  }
  __syncthreads();
#pragma unroll
  for (int nt = 0; nt < 4; ++nt)
#pragma unroll
    for (int r = 0; r < 8; ++r)
      hs[(r + hf * 8) * 64 + nt * 16 + l16] = (_Float16)acc[nt][r];
  __syncthreads();

  // h3 as A operands for the 64->320 layer
  v16h a0 = load_a_lds(hs + l16 * 64 + 0 + hf * 8);
  v16h a1 = load_a_lds(hs + l16 * 64 + 32 + hf * 8);

  // per-edge metadata; this lane's D elements r cover edge row r + hf*8
  int ii[8], jj[8];
  float eas[8], ev0[8], ev1[8], ev2[8];
#pragma unroll
  for (int r = 0; r < 8; ++r) {
    int er = e0 + r + hf * 8;
    ii[r] = iedges[er];
    jj[r] = jedges[er];
    const float* ea = edge_attrs + (size_t)er * 4;
    eas[r] = ea[0];
    ev0[r] = ea[1];
    ev1[r] = ea[2];
    ev2[r] = ea[3];
    __builtin_prefetch(s_up + (size_t)ii[r] * 64, 0, 0);
    __builtin_prefetch(v_up + (size_t)ii[r] * 192, 0, 0);
  }

  // ---- layer 4 (64 -> 320), fused message build + scatter per 16-col tile
#pragma unroll
  for (int nt = 0; nt < 20; ++nt) {
    v8f c = vzero8();
    c = wmma16(a0, load_b_lds(w4, nt, 2, 0, lane), c);
    c = wmma16(a1, load_b_lds(w4, nt, 2, 1, lane), c);
    const int chunk = nt >> 2;          // 0:wa 1:wb 2:wc 3:wd 4:we
    const int cc = (nt & 3) * 16 + l16; // channel 0..63 within chunk
#pragma unroll
    for (int r = 0; r < 8; ++r) {
      const float w = c[r];
      const int i = ii[r];
      const int j = jj[r];
      if (chunk == 0) {
        float xs = s_up[(size_t)i * 64 + cc];
        atomicAdd(&msg_s[(size_t)j * 128 + cc], w * xs * eas[r]);
      } else if (chunk == 1) {
        const float* xv = v_up + (size_t)i * 192 + cc * 3;
        float dot = xv[0] * ev0[r] + xv[1] * ev1[r] + xv[2] * ev2[r];
        atomicAdd(&msg_s[(size_t)j * 128 + 64 + cc], w * dot * INV_SQRT3_F);
      } else if (chunk == 2) {
        float t = w * s_up[(size_t)i * 64 + cc];
        float* d = msg_v + ((size_t)j * 192 + cc) * 3;
        atomicAdd(d + 0, t * ev0[r]);
        atomicAdd(d + 1, t * ev1[r]);
        atomicAdd(d + 2, t * ev2[r]);
      } else if (chunk == 3) {
        const float* xv = v_up + (size_t)i * 192 + cc * 3;
        float t = w * eas[r];
        float* d = msg_v + ((size_t)j * 192 + 64 + cc) * 3;
        atomicAdd(d + 0, t * xv[0]);
        atomicAdd(d + 1, t * xv[1]);
        atomicAdd(d + 2, t * xv[2]);
      } else {
        const float* xv = v_up + (size_t)i * 192 + cc * 3;
        float cr0 = xv[1] * ev2[r] - xv[2] * ev1[r];
        float cr1 = xv[2] * ev0[r] - xv[0] * ev2[r];
        float cr2 = xv[0] * ev1[r] - xv[1] * ev0[r];
        float t = w * INV_SQRT2_F;
        float* d = msg_v + ((size_t)j * 192 + 128 + cc) * 3;
        atomicAdd(d + 0, t * cr0);
        atomicAdd(d + 1, t * cr1);
        atomicAdd(d + 2, t * cr2);
      }
    }
  }
}

// ---------------------------------------------------------------------------
// Kernel 3: s2 = msg_s @ Wl0/sqrt(2C) ; v2[.,k,m] = msg_v[.,c,m] @ Wl1/sqrt(3C)
// ---------------------------------------------------------------------------
__global__ void __launch_bounds__(256) node_lin_kernel(
    const float* __restrict__ msg_s, const float* __restrict__ msg_v,
    const _Float16* __restrict__ wl0, const _Float16* __restrict__ wl1,
    float* __restrict__ s2, float* __restrict__ v2) {
  const int tid = threadIdx.x;
  const int wv = tid >> 5, lane = tid & 31;
  const int hf = lane >> 4, l16 = lane & 15;
  const int t = blockIdx.x * 8 + wv;
  if (t >= NNODES / 16) return;
  const int row0 = t * 16;
  const int myrow = row0 + l16;

  {  // s2: K = 128
    v16h as[4];
#pragma unroll
    for (int ks = 0; ks < 4; ++ks)
      as[ks] = load_a_f32(msg_s + (size_t)myrow * 128 + ks * 32 + hf * 8);
#pragma unroll
    for (int nt = 0; nt < 4; ++nt) {
      v8f c = vzero8();
#pragma unroll
      for (int ks = 0; ks < 4; ++ks)
        c = wmma16(as[ks], load_b_sw(wl0, nt, 4, ks, lane), c);
#pragma unroll
      for (int r = 0; r < 8; ++r)
        s2[(size_t)(row0 + r + hf * 8) * 64 + nt * 16 + l16] = c[r];
    }
  }
  // v2: K = 192, per component m
#pragma unroll
  for (int m = 0; m < 3; ++m) {
    v16h av[6];
#pragma unroll
    for (int ks = 0; ks < 6; ++ks)
      av[ks] = load_a_f32_s3(msg_v + ((size_t)myrow * 192 + ks * 32 + hf * 8) * 3 + m);
#pragma unroll
    for (int nt = 0; nt < 4; ++nt) {
      v8f c = vzero8();
#pragma unroll
      for (int ks = 0; ks < 6; ++ks)
        c = wmma16(av[ks], load_b_sw(wl1, nt, 6, ks, lane), c);
#pragma unroll
      for (int r = 0; r < 8; ++r)
        v2[(size_t)(row0 + r + hf * 8) * 192 + (nt * 16 + l16) * 3 + m] = c[r];
    }
  }
}

// ---------------------------------------------------------------------------
// Kernel 4: per-(node,channel) polynomial readout with Z-indexed weights.
// ---------------------------------------------------------------------------
__global__ void __launch_bounds__(256) pointwise_kernel(
    const float* __restrict__ s2, const float* __restrict__ v2,
    const int* __restrict__ anum, const float* __restrict__ Ws0c1,
    const float* __restrict__ Ws0c2, const float* __restrict__ Ws0c3,
    const float* __restrict__ Ws1c1, const float* __restrict__ Ws1c2,
    const float* __restrict__ Ws1c3, float* __restrict__ out_s,
    float* __restrict__ out_v) {
  int idx = blockIdx.x * blockDim.x + threadIdx.x;  // node*64 + c
  if (idx >= NNODES * 64) return;
  int n = idx >> 6, c = idx & 63;
  int z = anum[n];
  float s = s2[idx];
  const float* v = v2 + (size_t)idx * 3;
  float va = v[0], vb = v[1], vc = v[2];
  float vvs = (va * va + vb * vb + vc * vc) * INV_SQRT3_F;  // vv/sqrt(3)
  float w01  = Ws0c1[z * 64 + c];
  float w02a = Ws0c2[(z * 2 + 0) * 64 + c];
  float w02b = Ws0c2[(z * 2 + 1) * 64 + c];
  float w03a = Ws0c3[(z * 2 + 0) * 64 + c];
  float w03b = Ws0c3[(z * 2 + 1) * 64 + c];
  float w11  = Ws1c1[z * 64 + c];
  float w12  = Ws1c2[z * 64 + c];
  float w13a = Ws1c3[(z * 2 + 0) * 64 + c];
  float w13b = Ws1c3[(z * 2 + 1) * 64 + c];
  float sq = s * s;
  out_s[idx] = w01 * s + w02a * sq + w02b * vvs + w03a * sq * s + w03b * s * vvs;
  float coef = w11 + w12 * s + w13a * sq + w13b * vvs;
  float* ov = out_v + (size_t)idx * 3;
  ov[0] = coef * va;
  ov[1] = coef * vb;
  ov[2] = coef * vc;
}

// ---------------------------------------------------------------------------
// Kernel 5: output projections -> d_out (N x 256): [so | vo.reshape(192)]
// ---------------------------------------------------------------------------
__global__ void __launch_bounds__(256) final_kernel(
    const float* __restrict__ out_s, const float* __restrict__ out_v,
    const _Float16* __restrict__ wp0, const _Float16* __restrict__ wp1,
    float* __restrict__ out) {
  const int tid = threadIdx.x;
  const int wv = tid >> 5, lane = tid & 31;
  const int hf = lane >> 4, l16 = lane & 15;
  const int t = blockIdx.x * 8 + wv;
  if (t >= NNODES / 16) return;
  const int row0 = t * 16;
  const int myrow = row0 + l16;

  {  // so
    v16h a0 = load_a_f32(out_s + (size_t)myrow * 64 + 0 + hf * 8);
    v16h a1 = load_a_f32(out_s + (size_t)myrow * 64 + 32 + hf * 8);
#pragma unroll
    for (int nt = 0; nt < 4; ++nt) {
      v8f c = vzero8();
      c = wmma16(a0, load_b_sw(wp0, nt, 2, 0, lane), c);
      c = wmma16(a1, load_b_sw(wp0, nt, 2, 1, lane), c);
#pragma unroll
      for (int r = 0; r < 8; ++r)
        out[(size_t)(row0 + r + hf * 8) * 256 + nt * 16 + l16] = c[r];
    }
  }
#pragma unroll
  for (int m = 0; m < 3; ++m) {  // vo
    const float* rp = out_v + (size_t)myrow * 192 + m;
    v16h a0 = load_a_f32_s3(rp + 3 * (0 + hf * 8));
    v16h a1 = load_a_f32_s3(rp + 3 * (32 + hf * 8));
#pragma unroll
    for (int nt = 0; nt < 4; ++nt) {
      v8f c = vzero8();
      c = wmma16(a0, load_b_sw(wp1, nt, 2, 0, lane), c);
      c = wmma16(a1, load_b_sw(wp1, nt, 2, 1, lane), c);
#pragma unroll
      for (int r = 0; r < 8; ++r)
        out[(size_t)(row0 + r + hf * 8) * 256 + 64 + (nt * 16 + l16) * 3 + m] = c[r];
    }
  }
}

// ---------------------------------------------------------------------------
extern "C" void kernel_launch(void* const* d_in, const int* in_sizes, int n_in,
                              void* d_out, int out_size, void* d_ws,
                              size_t ws_size, hipStream_t stream) {
  (void)in_sizes; (void)n_in; (void)out_size; (void)ws_size;
  const float* node_feats = (const float*)d_in[0];
  const float* edge_feats = (const float*)d_in[1];
  const float* edge_attrs = (const float*)d_in[2];
  const int* iedges = (const int*)d_in[3];
  const int* jedges = (const int*)d_in[4];
  const int* anum   = (const int*)d_in[5];
  const float* Wup0 = (const float*)d_in[6];
  const float* Wup1 = (const float*)d_in[7];
  const float* Wm1  = (const float*)d_in[8];
  const float* Wm2  = (const float*)d_in[9];
  const float* Wm3  = (const float*)d_in[10];
  const float* Wm4  = (const float*)d_in[11];
  const float* Wl0  = (const float*)d_in[12];
  const float* Wl1  = (const float*)d_in[13];
  const float* Ws0c1 = (const float*)d_in[14];
  const float* Ws0c2 = (const float*)d_in[15];
  const float* Ws0c3 = (const float*)d_in[16];
  const float* Ws1c1 = (const float*)d_in[17];
  const float* Ws1c2 = (const float*)d_in[18];
  const float* Ws1c3 = (const float*)d_in[19];
  const float* Wp0  = (const float*)d_in[20];
  const float* Wp1  = (const float*)d_in[21];

  const size_t N = NNODES;
  float* ws = (float*)d_ws;
  float* s_up  = ws;                   // N*64   (reused as out_s)
  float* v_up  = s_up + N * 64;        // N*192  (reused as out_v)
  float* msg_s = v_up + N * 192;       // N*128
  float* msg_v = msg_s + N * 128;      // N*576
  float* s2    = msg_v + N * 576;      // N*64
  float* v2    = s2 + N * 64;          // N*192
  _Float16* hw = (_Float16*)(v2 + N * 192);  // 32B-aligned (offset % 32 == 0)
  _Float16* hw_up0 = hw;               // 64x64   -> 4096
  _Float16* hw_up1 = hw_up0 + 4096;    // 64x64   -> 4096
  _Float16* hw_m1  = hw_up1 + 4096;    // 8(->32)x64 -> 2048  | contiguous
  _Float16* hw_m2  = hw_m1 + 2048;     // 64x64   -> 4096     | 61440-byte
  _Float16* hw_m3  = hw_m2 + 4096;     // 64x64   -> 4096     | region for
  _Float16* hw_m4  = hw_m3 + 4096;     // 64x320  -> 20480    | one TDM load
  _Float16* hw_l0  = hw_m4 + 20480;    // 128x64  -> 8192
  _Float16* hw_l1  = hw_l0 + 8192;     // 192x64  -> 12288
  _Float16* hw_p0  = hw_l1 + 12288;    // 64x64   -> 4096
  _Float16* hw_p1  = hw_p0 + 4096;     // 64x64   -> 4096

  // 1) pre-swizzle all weights into WMMA B-operand layout, scales folded
  swizzle_w_kernel<<<64, 256, 0, stream>>>(Wup0, hw_up0, 64, 64, 64, INV_SQRT_C_F);
  swizzle_w_kernel<<<64, 256, 0, stream>>>(Wup1, hw_up1, 64, 64, 64, INV_SQRT_C_F);
  swizzle_w_kernel<<<64, 256, 0, stream>>>(Wm1,  hw_m1,   8, 32, 64, INV_SQRT_NB_F);
  swizzle_w_kernel<<<64, 256, 0, stream>>>(Wm2,  hw_m2,  64, 64, 64, 0.125f);
  swizzle_w_kernel<<<64, 256, 0, stream>>>(Wm3,  hw_m3,  64, 64, 64, 0.125f);
  swizzle_w_kernel<<<160, 256, 0, stream>>>(Wm4, hw_m4,  64, 64, 320, 0.125f);
  swizzle_w_kernel<<<64, 256, 0, stream>>>(Wl0,  hw_l0, 128, 128, 64, INV_SQRT_2C_F);
  swizzle_w_kernel<<<96, 256, 0, stream>>>(Wl1,  hw_l1, 192, 192, 64, INV_SQRT_3C_F);
  swizzle_w_kernel<<<64, 256, 0, stream>>>(Wp0,  hw_p0,  64, 64, 64, INV_SQRT_C_F);
  swizzle_w_kernel<<<64, 256, 0, stream>>>(Wp1,  hw_p1,  64, 64, 64, INV_SQRT_C_F);

  // 2) zero message accumulators (msg_s and msg_v are contiguous)
  zero_kernel<<<2048, 256, 0, stream>>>(msg_s, N * (128 + 576));

  // 3) node up-projection (1250 row tiles, 8 waves/block)
  node_up_kernel<<<157, 256, 0, stream>>>(node_feats, hw_up0, hw_up1, s_up, v_up);

  // 4) fused edge MLP + message + scatter (20000 edge tiles); TDM weight stage
  edge_kernel<<<2500, 256, EDGE_LDS_BYTES, stream>>>(
      edge_feats, edge_attrs, iedges, jedges, hw_m1, s_up, v_up, msg_s, msg_v);

  // 5) node linear layers
  node_lin_kernel<<<157, 256, 0, stream>>>(msg_s, msg_v, hw_l0, hw_l1, s2, v2);

  // 6) pointwise polynomial readout (out_s/out_v alias s_up/v_up: edge_kernel
  //    has already consumed them, stream ordering guarantees safety)
  float* out_s = s_up;
  float* out_v = v_up;
  pointwise_kernel<<<5000, 256, 0, stream>>>(s2, v2, anum, Ws0c1, Ws0c2, Ws0c3,
                                             Ws1c1, Ws1c2, Ws1c3, out_s, out_v);

  // 7) output projections -> d_out (N x 256 f32)
  final_kernel<<<157, 256, 0, stream>>>(out_s, out_v, hw_p0, hw_p1,
                                        (float*)d_out);
}
